// ScaledDotProductAttention_54348516164079
// MI455X (gfx1250) — compile-verified, matrix operand order checked by reading, not run on hardware
//
#include <hip/hip_runtime.h>
#include <hip/hip_bf16.h>

typedef __bf16 v16bf __attribute__((ext_vector_type(16)));
typedef float  v8f   __attribute__((ext_vector_type(8)));

#define B_ 2
#define H_ 16
#define S_ 2048
#define D_ 128
#define QT 32            // query rows per workgroup
#define KT 64            // key rows per tile
#define NT (S_ / KT)     // 32 key tiles
#define PST 2064         // score panel row stride (bf16 elems), padded vs 2048
#define QS_STRIDE 136    // 128 + 8 pad
#define KS_STRIDE 136
#define VT_STRIDE 72     // 64 + 8 pad

// LDS byte offsets (total 213760 < 320KB per WGP)
#define OFF_PANEL 0
#define SZ_PANEL  (QT * PST * 2)                    // 132096
#define OFF_QS    (OFF_PANEL + SZ_PANEL)            // 132096
#define SZ_QS     (QT * QS_STRIDE * 2)              // 8704
#define OFF_KS0   (OFF_QS + SZ_QS)                  // 140800
#define SZ_KS     (KT * KS_STRIDE * 2)              // 17408
#define OFF_KS1   (OFF_KS0 + SZ_KS)                 // 158208
#define OFF_VT0   (OFF_KS1 + SZ_KS)                 // 175616
#define SZ_VT     (D_ * VT_STRIDE * 2)              // 18432
#define OFF_VT1   (OFF_VT0 + SZ_VT)                 // 194048
#define OFF_PART  (OFF_VT1 + SZ_VT)                 // 212480
#define OFF_RMAX  (OFF_PART + 256 * 4)              // 213504
#define OFF_RINV  (OFF_RMAX + 32 * 4)               // 213632
#define LDS_TOTAL (OFF_RINV + 32 * 4)               // 213760

// 16x32 bf16 operand tile (ISA 16-bit A layout; B identical when the source
// array is [N][K] row-major): lane<16 holds K={0..7,16..23}, lane>=16 holds
// K={8..15,24..31}, row = lane&15.
__device__ __forceinline__ v16bf load_ab(const __bf16* base, int stride,
                                         int row_base, int k_base) {
  const int lane = threadIdx.x & 31;
  const int lo = lane & 15, hi = lane >> 4;
  const __bf16* p = base + (row_base + lo) * stride + k_base + hi * 8;
  union { v16bf v; uint4 u[2]; } t;
  t.u[0] = *reinterpret_cast<const uint4*>(p);
  t.u[1] = *reinterpret_cast<const uint4*>(p + 16);
  return t.v;
}

__device__ __forceinline__ v8f wmma_bf16(v16bf a, v16bf b, v8f c) {
  return __builtin_amdgcn_wmma_f32_16x16x32_bf16(false, a, false, b,
                                                 (short)0, c, false, false);
}

union pk4 { __bf16 b[4]; uint2 u; };
union pk8 { __bf16 b[8]; uint4 u; };

__global__ __launch_bounds__(256, 1)
void sdpa_wmma_kernel(const float* __restrict__ Q, const float* __restrict__ K,
                      const float* __restrict__ V, float* __restrict__ Out,
                      float* __restrict__ W) {
  extern __shared__ char smem[];
  __bf16* panel = reinterpret_cast<__bf16*>(smem + OFF_PANEL);
  __bf16* qs    = reinterpret_cast<__bf16*>(smem + OFF_QS);
  __bf16* ks0   = reinterpret_cast<__bf16*>(smem + OFF_KS0);
  __bf16* ks1   = reinterpret_cast<__bf16*>(smem + OFF_KS1);
  __bf16* vt0   = reinterpret_cast<__bf16*>(smem + OFF_VT0);
  __bf16* vt1   = reinterpret_cast<__bf16*>(smem + OFF_VT1);
  float*  part   = reinterpret_cast<float*>(smem + OFF_PART);
  float*  rowmax = reinterpret_cast<float*>(smem + OFF_RMAX);
  float*  rowinv = reinterpret_cast<float*>(smem + OFF_RINV);

  const int tid  = threadIdx.x;
  const int bh   = blockIdx.y;
  const int q0   = blockIdx.x * QT;
  const int wv   = tid >> 5;
  const int lane = tid & 31;
  const int lo   = lane & 15, hi = lane >> 4;

  const size_t bh_base = (size_t)bh * S_ * D_;
  const float* Kbase = K + bh_base;
  const float* Vbase = V + bh_base;

  // ---- stage Q tile (32x128 f32 -> bf16 LDS) ----
  #pragma unroll
  for (int i = 0; i < 4; ++i) {
    int e = tid + i * 256, row = e >> 5, c4 = (e & 31) * 4;
    const float4 f = *reinterpret_cast<const float4*>(
        Q + bh_base + (size_t)(q0 + row) * D_ + c4);
    pk4 pk; pk.b[0] = (__bf16)f.x; pk.b[1] = (__bf16)f.y;
    pk.b[2] = (__bf16)f.z; pk.b[3] = (__bf16)f.w;
    *reinterpret_cast<uint2*>(qs + row * QS_STRIDE + c4) = pk.u;
  }
  // ---- prologue: stage K tile 0 ----
  #pragma unroll
  for (int i = 0; i < 8; ++i) {
    int e = tid + i * 256, row = e >> 5, c4 = (e & 31) * 4;
    const float4 f = *reinterpret_cast<const float4*>(
        Kbase + (size_t)row * D_ + c4);
    pk4 pk; pk.b[0] = (__bf16)f.x; pk.b[1] = (__bf16)f.y;
    pk.b[2] = (__bf16)f.z; pk.b[3] = (__bf16)f.w;
    *reinterpret_cast<uint2*>(ks0 + row * KS_STRIDE + c4) = pk.u;
  }
  __syncthreads();

  const int m_base = (wv >> 2) * 16;      // 2 M tiles x 4 N tiles over 32x64
  const int n_idx  = wv & 3;
  const float scale = 0.08838834764831845f;   // 1/sqrt(128)

  // Q A-operands are loop-invariant: hoist (4 x v16bf = 32 VGPRs)
  v16bf aq[4];
  #pragma unroll
  for (int kk = 0; kk < 4; ++kk) aq[kk] = load_ab(qs, QS_STRIDE, m_base, kk * 32);

  // ---- phase 1: scores = (Q K^T)/sqrt(D) into LDS panel (double-buffered) ----
  for (int kt = 0; kt < NT; ++kt) {
    __bf16* cur = (kt & 1) ? ks1 : ks0;
    __bf16* nxt = (kt & 1) ? ks0 : ks1;
    const bool have_next = (kt + 1 < NT);

    float4 f[8];                           // issue next tile's global loads now
    if (have_next) {
      const float* g = Kbase + (size_t)(kt + 1) * KT * D_;
      #pragma unroll
      for (int i = 0; i < 8; ++i) {
        int e = tid + i * 256;
        f[i] = *reinterpret_cast<const float4*>(g + (size_t)(e >> 5) * D_ + (e & 31) * 4);
      }
    }
    if (kt + 2 < NT)                       // pull tile kt+2 toward L2/WGP$
      __builtin_prefetch(Kbase + (size_t)(kt + 2) * KT * D_ + tid * 32, 0, 1);

    v8f c = {};
    #pragma unroll
    for (int kk = 0; kk < 4; ++kk) {
      v16bf b = load_ab(cur, KS_STRIDE, n_idx * 16, kk * 32);
      c = wmma_bf16(aq[kk], b, c);
    }
    #pragma unroll
    for (int r = 0; r < 8; ++r) {          // C/D layout: M=r+8*hi, N=lo
      int row = m_base + r + 8 * hi;
      int col = kt * KT + n_idx * 16 + lo;
      panel[row * PST + col] = (__bf16)(c[r] * scale);
    }
    if (have_next) {                       // convert + store next tile
      #pragma unroll
      for (int i = 0; i < 8; ++i) {
        int e = tid + i * 256, row = e >> 5, c4 = (e & 31) * 4;
        pk4 pk; pk.b[0] = (__bf16)f[i].x; pk.b[1] = (__bf16)f[i].y;
        pk.b[2] = (__bf16)f[i].z; pk.b[3] = (__bf16)f[i].w;
        *reinterpret_cast<uint2*>(nxt + row * KS_STRIDE + c4) = pk.u;
      }
    }
    __syncthreads();
  }

  // ---- softmax: row max (8 bf16 per ds_load_b128) ----
  {
    int r = tid >> 3, seg = tid & 7;
    const __bf16* prow = panel + r * PST + seg * 256;
    float mx = -3.0e38f;
    for (int i = 0; i < 32; ++i) {
      pk8 t; t.u = *reinterpret_cast<const uint4*>(prow + i * 8);
      #pragma unroll
      for (int j = 0; j < 8; ++j) mx = fmaxf(mx, (float)t.b[j]);
    }
    part[tid] = mx;
  }
  __syncthreads();
  if (tid < 32) {
    float mx = part[tid * 8];
    #pragma unroll
    for (int j = 1; j < 8; ++j) mx = fmaxf(mx, part[tid * 8 + j]);
    rowmax[tid] = mx;
  }
  __syncthreads();
  // ---- exp + sum (vectorized, overwrite panel with exp) ----
  {
    int r = tid >> 3, seg = tid & 7;
    __bf16* prow = panel + r * PST + seg * 256;
    const float m = rowmax[r];
    float sum = 0.f;
    for (int i = 0; i < 32; ++i) {
      pk8 t; t.u = *reinterpret_cast<const uint4*>(prow + i * 8);
      #pragma unroll
      for (int j = 0; j < 8; ++j) {
        float e = __expf((float)t.b[j] - m);
        t.b[j] = (__bf16)e;
        sum += e;
      }
      *reinterpret_cast<uint4*>(prow + i * 8) = t.u;
    }
    part[tid] = sum;
  }
  __syncthreads();
  if (tid < 32) {
    float s = 0.f;
    #pragma unroll
    for (int j = 0; j < 8; ++j) s += part[tid * 8 + j];
    rowinv[tid] = 1.0f / s;
  }
  __syncthreads();
  // ---- normalize: single coalesced write of weights (2x b128 per iter) ----
  {
    float* wbase = W + ((size_t)bh * S_ + q0) * S_;
    for (int g8 = tid; g8 < QT * S_ / 8; g8 += 256) {
      int r = g8 >> 8;                    // 256 groups of 8 per 2048-wide row
      int k = (g8 & 255) * 8;
      const float inv = rowinv[r];
      pk8 t; t.u = *reinterpret_cast<const uint4*>(panel + r * PST + k);
      float w0[8];
      #pragma unroll
      for (int j = 0; j < 8; ++j) { w0[j] = (float)t.b[j] * inv; t.b[j] = (__bf16)w0[j]; }
      float4* wp = reinterpret_cast<float4*>(wbase + (size_t)r * S_ + k);
      wp[0] = make_float4(w0[0], w0[1], w0[2], w0[3]);
      wp[1] = make_float4(w0[4], w0[5], w0[6], w0[7]);
      *reinterpret_cast<uint4*>(panel + r * PST + k) = t.u;   // normalized P
    }
  }

  // ---- prologue: stage V tile 0 transposed vt0[d][k] ----
  #pragma unroll
  for (int i = 0; i < 8; ++i) {
    int e = tid + i * 256, row = e >> 5, c4 = (e & 31) * 4;
    const float4 f = *reinterpret_cast<const float4*>(
        Vbase + (size_t)row * D_ + c4);
    vt0[(c4 + 0) * VT_STRIDE + row] = (__bf16)f.x;
    vt0[(c4 + 1) * VT_STRIDE + row] = (__bf16)f.y;
    vt0[(c4 + 2) * VT_STRIDE + row] = (__bf16)f.z;
    vt0[(c4 + 3) * VT_STRIDE + row] = (__bf16)f.w;
  }
  __syncthreads();   // covers panel normalization + vt0 staging

  // ---- phase 2: O = P V (double-buffered V) ----
  const int d_base = (wv & 3) * 16;       // wave covers D cols d_base, d_base+64
  v8f o0 = {}, o1 = {};
  for (int kt = 0; kt < NT; ++kt) {
    __bf16* cur = (kt & 1) ? vt1 : vt0;
    __bf16* nxt = (kt & 1) ? vt0 : vt1;
    const bool have_next = (kt + 1 < NT);

    float4 f[8];
    if (have_next) {
      const float* g = Vbase + (size_t)(kt + 1) * KT * D_;
      #pragma unroll
      for (int i = 0; i < 8; ++i) {
        int e = tid + i * 256;
        f[i] = *reinterpret_cast<const float4*>(g + (size_t)(e >> 5) * D_ + (e & 31) * 4);
      }
    }
    if (kt + 2 < NT)
      __builtin_prefetch(Vbase + (size_t)(kt + 2) * KT * D_ + tid * 32, 0, 1);

    #pragma unroll
    for (int kk = 0; kk < 2; ++kk) {      // 64 keys = 2x K=32
      v16bf a  = load_ab(panel, PST, m_base, kt * KT + kk * 32);
      v16bf b0 = load_ab(cur, VT_STRIDE, d_base, kk * 32);
      v16bf b1 = load_ab(cur, VT_STRIDE, d_base + 64, kk * 32);
      o0 = wmma_bf16(a, b0, o0);
      o1 = wmma_bf16(a, b1, o1);
    }
    if (have_next) {
      #pragma unroll
      for (int i = 0; i < 8; ++i) {
        int e = tid + i * 256, row = e >> 5, c4 = (e & 31) * 4;
        nxt[(c4 + 0) * VT_STRIDE + row] = (__bf16)f[i].x;
        nxt[(c4 + 1) * VT_STRIDE + row] = (__bf16)f[i].y;
        nxt[(c4 + 2) * VT_STRIDE + row] = (__bf16)f[i].z;
        nxt[(c4 + 3) * VT_STRIDE + row] = (__bf16)f[i].w;
      }
    }
    __syncthreads();
  }

  // ---- write O (f32) per C/D layout ----
  #pragma unroll
  for (int r = 0; r < 8; ++r) {
    int row = m_base + r + 8 * hi;
    size_t o = bh_base + (size_t)(q0 + row) * D_;
    Out[o + d_base + lo]      = o0[r];
    Out[o + d_base + 64 + lo] = o1[r];
  }
}

extern "C" void kernel_launch(void* const* d_in, const int* in_sizes, int n_in,
                              void* d_out, int out_size, void* d_ws, size_t ws_size,
                              hipStream_t stream) {
  const float* Q = (const float*)d_in[0];
  const float* K = (const float*)d_in[1];
  const float* V = (const float*)d_in[2];
  float* Out = (float*)d_out;
  float* W   = Out + (size_t)B_ * H_ * S_ * D_;   // weights follow output

  dim3 grid(S_ / QT, B_ * H_);
  dim3 block(256);
  sdpa_wmma_kernel<<<grid, block, LDS_TOTAL, stream>>>(Q, K, V, Out, W);
}
